// MultiHeadSelfAttention_32049045962835
// MI455X (gfx1250) — matrix operand order repaired and checked by reading, not compiled, not run on hardware
//
#include <hip/hip_runtime.h>

#define DM   1024
#define NH   16
#define DK   64
#define BSZ  2
#define SEQ  2048
#define MROWS (BSZ * SEQ)   // 4096

typedef __attribute__((ext_vector_type(16))) __bf16  v16bf;
typedef __attribute__((ext_vector_type(8)))  float   v8f;
typedef __attribute__((ext_vector_type(4)))  unsigned int u32x4;
typedef __attribute__((ext_vector_type(4)))  int i32x4;

union FragU { v16bf v; u32x4 q[2]; unsigned short u[16]; };

#if defined(__has_builtin)
# if __has_builtin(__builtin_amdgcn_global_load_async_to_lds_b128)
#  define HAVE_ASYNC_LDS 1
# endif
#endif
#ifndef HAVE_ASYNC_LDS
# define HAVE_ASYNC_LDS 0
#endif

__device__ __forceinline__ unsigned short f2bf(float f) {
    unsigned int u = __float_as_uint(f);
    u += 0x7FFFu + ((u >> 16) & 1u);   // round-to-nearest-even
    return (unsigned short)(u >> 16);
}

__device__ __forceinline__ v8f wmma_bf16(v16bf a, v16bf b, v8f c) {
    return __builtin_amdgcn_wmma_f32_16x16x32_bf16(
        false, a, false, b, (short)0, c, false, false);
}

// A-operand fragment (16x32 MxK, bf16). Source row-major, leading dim ld (halves).
// lane<16: K = k0+[0..7] and k0+16+[0..7]; lane>=16: same +8.
__device__ __forceinline__ v16bf load_a(const unsigned short* base, int ld,
                                        int row, int k0, int lane) {
    const unsigned short* p = base + row * ld + k0 + ((lane >> 4) << 3);
    FragU f;
    f.q[0] = *(const u32x4*)(p);
    f.q[1] = *(const u32x4*)(p + 16);
    return f.v;
}

// B-operand fragment (32x16 KxN, bf16). Column n comes from row `col` of a
// row-major matrix with leading dim ld (i.e. B = that matrix transposed).
// lane<16: K = k0+[0..7] and k0+16+[0..7]; lane>=16: same +8.
__device__ __forceinline__ v16bf load_b(const unsigned short* base, int ld,
                                        int col, int k0, int lane) {
    const unsigned short* p = base + col * ld + k0 + ((lane >> 4) << 3);
    FragU f;
    f.q[0] = *(const u32x4*)(p);
    f.q[1] = *(const u32x4*)(p + 16);
    return f.v;
}

// 16-byte global -> LDS copy, async DMA path when available.
__device__ __forceinline__ void cp16_async(const unsigned short* g, unsigned short* l) {
#if HAVE_ASYNC_LDS
    __builtin_amdgcn_global_load_async_to_lds_b128(
        (__attribute__((address_space(1))) i32x4*)g,
        (__attribute__((address_space(3))) i32x4*)l, 0, 0);
#else
    *(u32x4*)l = *(const u32x4*)g;
#endif
}

__device__ __forceinline__ void wait_async0() {
#if HAVE_ASYNC_LDS
# if __has_builtin(__builtin_amdgcn_s_wait_asynccnt)
    __builtin_amdgcn_s_wait_asynccnt(0);
# else
    asm volatile("s_wait_asynccnt 0x0" ::: "memory");
# endif
#endif
}

// ---------------------------------------------------------------- convert
__global__ void cvt_f32_bf16(const float* __restrict__ src,
                             unsigned short* __restrict__ dst, int n) {
    int i = blockIdx.x * blockDim.x + threadIdx.x;
    int stride = gridDim.x * blockDim.x;
    for (; i < n; i += stride) dst[i] = f2bf(src[i]);
}

// ---------------------------------------------------------------- QKV proj
// grid = (N/128, M/128, 3), block = 256 (8 waves, 2M x 4N).
// Each wave owns a 64x32 tile: 8 WMMA per 12 b128 loads per k-step.
// mode 0: Q -> RoPE -> (b,h,s,d);  mode 1: K -> RoPE -> (b,h,s,d);
// mode 2: V -> (b,h,d,s) transposed.
__global__ __launch_bounds__(256) void proj_qkv(
    const unsigned short* __restrict__ Xb,
    const unsigned short* __restrict__ Wqb,
    const unsigned short* __restrict__ Wkb,
    const unsigned short* __restrict__ Wvb,
    unsigned short* __restrict__ Qh,
    unsigned short* __restrict__ Kh,
    unsigned short* __restrict__ Vt)
{
    const int mode = blockIdx.z;
    const unsigned short* W   = (mode == 0) ? Wqb : (mode == 1) ? Wkb : Wvb;
    unsigned short*       Out = (mode == 0) ? Qh : (mode == 1) ? Kh : Vt;

    const int lane = threadIdx.x & 31;
    const int wid  = threadIdx.x >> 5;
    const int wm   = wid & 1;      // 2 waves along M (64 rows each)
    const int wn   = wid >> 1;     // 4 waves along N (32 cols each)
    const int bm   = blockIdx.y * 128;
    const int bn   = blockIdx.x * 128;

    const int r0 = bm + wm * 64 + (lane & 15);
    const int c0 = bn + wn * 32 + (lane & 15);

    v8f acc[4][2] = {};
    for (int k0 = 0; k0 < DM; k0 += 32) {
        v16bf a0 = load_a(Xb, DM, r0,      k0, lane);
        v16bf a1 = load_a(Xb, DM, r0 + 16, k0, lane);
        v16bf a2 = load_a(Xb, DM, r0 + 32, k0, lane);
        v16bf a3 = load_a(Xb, DM, r0 + 48, k0, lane);
        v16bf b0 = load_b(W,  DM, c0,      k0, lane);
        v16bf b1 = load_b(W,  DM, c0 + 16, k0, lane);
        acc[0][0] = wmma_bf16(a0, b0, acc[0][0]);
        acc[0][1] = wmma_bf16(a0, b1, acc[0][1]);
        acc[1][0] = wmma_bf16(a1, b0, acc[1][0]);
        acc[1][1] = wmma_bf16(a1, b1, acc[1][1]);
        acc[2][0] = wmma_bf16(a2, b0, acc[2][0]);
        acc[2][1] = wmma_bf16(a2, b1, acc[2][1]);
        acc[3][0] = wmma_bf16(a3, b0, acc[3][0]);
        acc[3][1] = wmma_bf16(a3, b1, acc[3][1]);
    }

    const float LN_THETA_OVER_DK = 9.210340371976184f / 64.0f; // ln(10000)/64
    #pragma unroll
    for (int mt = 0; mt < 4; ++mt) {
        #pragma unroll
        for (int nt = 0; nt < 2; ++nt) {
            const int n = bn + wn * 32 + nt * 16 + (lane & 15);
            const int h = n >> 6, d = n & 63;
            #pragma unroll
            for (int i = 0; i < 8; ++i) {
                const int m = bm + wm * 64 + mt * 16 + i + ((lane >> 4) << 3);
                const int b = m >> 11, s = m & (SEQ - 1);
                float v = acc[mt][nt][i];
                if (mode < 2) {
                    // RoPE: even/odd dim pair lives in lane pair (d ^ 1 <-> lane ^ 1)
                    float partner = __shfl_xor(v, 1, 32);
                    float inv = __expf(-(float)(d & ~1) * LN_THETA_OVER_DK);
                    float sn, cs;
                    __sincosf((float)s * inv, &sn, &cs);
                    float r = (d & 1) ? (partner * sn + v * cs)
                                      : (v * cs - partner * sn);
                    Out[(((b << 4) + h) * SEQ + s) * DK + d] = f2bf(r);
                } else {
                    Out[(((b << 4) + h) * DK + d) * SEQ + s] = f2bf(v);
                }
            }
        }
    }
}

// ---------------------------------------------------------------- attention
// grid = (SEQ/64, B*NH), block = 128 (4 waves). Each block: 64 query rows.
// K/V 64x64 tiles are staged into LDS with async-to-LDS DMA, double-buffered:
// block j+1 is prefetched while block j is computed. Softmax statistics are
// register-resident per wave (shfl_xor reductions within the 16-lane half).
__global__ __launch_bounds__(128) void attn(
    const unsigned short* __restrict__ Qh,
    const unsigned short* __restrict__ Kh,
    const unsigned short* __restrict__ Vt,
    unsigned short* __restrict__ Ctx)
{
    __shared__ alignas(16) unsigned short Kbuf[2][64][64];   // (key, d)
    __shared__ alignas(16) unsigned short Vbuf[2][64][64];   // (d, key-local)
    __shared__ alignas(16) unsigned short Pw[4][16][64];     // per-wave P tile

    const int tid  = threadIdx.x;
    const int lane = tid & 31;
    const int wid  = tid >> 5;                // 0..3, 16-row band each
    const int qb   = blockIdx.x;              // query block (64 rows)
    const int bh   = blockIdx.y;
    const int b    = bh >> 4, h = bh & 15;

    const unsigned short* Qbase = Qh + (bh * SEQ + qb * 64) * DK;
    const unsigned short* Kh0   = Kh + bh * SEQ * DK;
    const unsigned short* Vbase = Vt + bh * DK * SEQ;

    // cooperative 64x64 K + V tile staging: 512 16B chunks each, 4 per thread
    auto stage_kv = [&](int jb2, int buf) {
        #pragma unroll
        for (int c = 0; c < 4; ++c) {
            const int id  = tid + 128 * c;
            const int row = id >> 3;
            const int off = (id & 7) << 3;
            cp16_async(Kh0 + (jb2 * 64 + row) * DK + off, &Kbuf[buf][row][off]);
            cp16_async(Vbase + row * SEQ + jb2 * 64 + off, &Vbuf[buf][row][off]);
        }
    };

    const int rhalf = (lane >> 4) << 3;                 // 0 or 8
    const int rowl  = wid * 16 + (lane & 15);           // local row for Q A-frag
    v16bf aq0 = load_a(Qbase, DK, rowl, 0,  lane);
    v16bf aq1 = load_a(Qbase, DK, rowl, 32, lane);

    v8f O[4] = {};
    float mrow[8], lrow[8];
    #pragma unroll
    for (int i = 0; i < 8; ++i) { mrow[i] = -3.0e38f; lrow[i] = 0.f; }

    stage_kv(0, 0);
    wait_async0();
    __syncthreads();

    for (int jb = 0; jb <= qb; ++jb) {
        const int cur = jb & 1;
        if (jb < qb) stage_kv(jb + 1, 1 - cur);   // prefetch next tiles (async)

        // S = Q K^T * 1/sqrt(dk); causal mask on the diagonal block
        v8f sc[4];
        #pragma unroll
        for (int nt = 0; nt < 4; ++nt) {
            const int col = nt * 16 + (lane & 15);
            v16bf bk0 = load_b(&Kbuf[cur][0][0], 64, col, 0,  lane);
            v16bf bk1 = load_b(&Kbuf[cur][0][0], 64, col, 32, lane);
            v8f c = {};
            c = wmma_bf16(aq0, bk0, c);
            c = wmma_bf16(aq1, bk1, c);
            c = c * 0.125f;
            if (jb == qb) {
                #pragma unroll
                for (int i = 0; i < 8; ++i)
                    if (col > wid * 16 + i + rhalf) c[i] = -3.0e30f;
            }
            sc[nt] = c;
        }

        // online softmax per row, fully in registers (rows are wave-private;
        // all 64 columns of row (i,half) live in sc[0..3][i] across the half)
        #pragma unroll
        for (int i = 0; i < 8; ++i) {
            float mx = mrow[i];
            #pragma unroll
            for (int nt = 0; nt < 4; ++nt) mx = fmaxf(mx, sc[nt][i]);
            mx = fmaxf(mx, __shfl_xor(mx, 1, 32));
            mx = fmaxf(mx, __shfl_xor(mx, 2, 32));
            mx = fmaxf(mx, __shfl_xor(mx, 4, 32));
            mx = fmaxf(mx, __shfl_xor(mx, 8, 32));
            float a = __expf(mrow[i] - mx);
            float sum = 0.f;
            #pragma unroll
            for (int nt = 0; nt < 4; ++nt) {
                float p = __expf(sc[nt][i] - mx);
                sc[nt][i] = p;
                sum += p;
            }
            sum += __shfl_xor(sum, 1, 32);
            sum += __shfl_xor(sum, 2, 32);
            sum += __shfl_xor(sum, 4, 32);
            sum += __shfl_xor(sum, 8, 32);
            lrow[i] = lrow[i] * a + sum;
            mrow[i] = mx;
            #pragma unroll
            for (int nt = 0; nt < 4; ++nt) O[nt][i] *= a;
        }

        // P (bf16) -> wave-private LDS tile, then reload as A-fragments
        #pragma unroll
        for (int nt = 0; nt < 4; ++nt)
            #pragma unroll
            for (int i = 0; i < 8; ++i)
                Pw[wid][i + rhalf][nt * 16 + (lane & 15)] = f2bf(sc[nt][i]);

        v16bf ap0 = load_a(&Pw[wid][0][0], 64, lane & 15, 0,  lane);
        v16bf ap1 = load_a(&Pw[wid][0][0], 64, lane & 15, 32, lane);

        // O += P V   (V^T tile: B-operand rows contiguous in LDS)
        #pragma unroll
        for (int nt = 0; nt < 4; ++nt) {
            const int d = nt * 16 + (lane & 15);
            v16bf bv0 = load_b(&Vbuf[cur][0][0], 64, d, 0,  lane);
            v16bf bv1 = load_b(&Vbuf[cur][0][0], 64, d, 32, lane);
            O[nt] = wmma_bf16(ap0, bv0, O[nt]);
            O[nt] = wmma_bf16(ap1, bv1, O[nt]);
        }

        if (jb < qb) {
            wait_async0();        // next tiles landed
            __syncthreads();      // all waves done with cur + prefetch visible
        }
    }

    #pragma unroll
    for (int nt = 0; nt < 4; ++nt) {
        const int d = nt * 16 + (lane & 15);
        #pragma unroll
        for (int i = 0; i < 8; ++i) {
            const int s = qb * 64 + wid * 16 + i + rhalf;
            Ctx[(b * SEQ + s) * DM + h * DK + d] = f2bf(O[nt][i] / lrow[i]);
        }
    }
}

// ---------------------------------------------------------------- out proj
__global__ __launch_bounds__(256) void out_proj(
    const unsigned short* __restrict__ Ctx,
    const unsigned short* __restrict__ Wob,
    float* __restrict__ Out)
{
    const int lane = threadIdx.x & 31;
    const int wid  = threadIdx.x >> 5;
    const int wm   = wid & 1;
    const int wn   = wid >> 1;
    const int bm   = blockIdx.y * 128;
    const int bn   = blockIdx.x * 128;

    const int r0 = bm + wm * 64 + (lane & 15);
    const int c0 = bn + wn * 32 + (lane & 15);

    v8f acc[4][2] = {};
    for (int k0 = 0; k0 < DM; k0 += 32) {
        v16bf a0 = load_a(Ctx, DM, r0,      k0, lane);
        v16bf a1 = load_a(Ctx, DM, r0 + 16, k0, lane);
        v16bf a2 = load_a(Ctx, DM, r0 + 32, k0, lane);
        v16bf a3 = load_a(Ctx, DM, r0 + 48, k0, lane);
        v16bf b0 = load_b(Wob, DM, c0,      k0, lane);
        v16bf b1 = load_b(Wob, DM, c0 + 16, k0, lane);
        acc[0][0] = wmma_bf16(a0, b0, acc[0][0]);
        acc[0][1] = wmma_bf16(a0, b1, acc[0][1]);
        acc[1][0] = wmma_bf16(a1, b0, acc[1][0]);
        acc[1][1] = wmma_bf16(a1, b1, acc[1][1]);
        acc[2][0] = wmma_bf16(a2, b0, acc[2][0]);
        acc[2][1] = wmma_bf16(a2, b1, acc[2][1]);
        acc[3][0] = wmma_bf16(a3, b0, acc[3][0]);
        acc[3][1] = wmma_bf16(a3, b1, acc[3][1]);
    }

    #pragma unroll
    for (int mt = 0; mt < 4; ++mt) {
        #pragma unroll
        for (int nt = 0; nt < 2; ++nt) {
            const int n = bn + wn * 32 + nt * 16 + (lane & 15);
            #pragma unroll
            for (int i = 0; i < 8; ++i) {
                const int m = bm + wm * 64 + mt * 16 + i + ((lane >> 4) << 3);
                Out[m * DM + n] = acc[mt][nt][i];
            }
        }
    }
}

// ---------------------------------------------------------------- launch
extern "C" void kernel_launch(void* const* d_in, const int* in_sizes, int n_in,
                              void* d_out, int out_size, void* d_ws, size_t ws_size,
                              hipStream_t stream) {
    const float* x  = (const float*)d_in[0];
    const float* Wq = (const float*)d_in[1];
    const float* Wk = (const float*)d_in[2];
    const float* Wv = (const float*)d_in[3];
    const float* Wo = (const float*)d_in[4];
    float* out = (float*)d_out;

    unsigned short* xb  = (unsigned short*)d_ws;       // 4096x1024
    unsigned short* wqb = xb  + MROWS * DM;            // 1024x1024 each
    unsigned short* wkb = wqb + DM * DM;
    unsigned short* wvb = wkb + DM * DM;
    unsigned short* wob = wvb + DM * DM;
    unsigned short* qh  = wob + DM * DM;               // (b,h,s,d)
    unsigned short* kh  = qh  + MROWS * DM;            // (b,h,s,d)
    unsigned short* vt  = kh  + MROWS * DM;            // (b,h,d,s)
    unsigned short* ctx = vt  + MROWS * DM;            // (b*s, 1024)

    cvt_f32_bf16<<<1024, 256, 0, stream>>>(x,  xb,  MROWS * DM);
    cvt_f32_bf16<<<512,  256, 0, stream>>>(Wq, wqb, DM * DM);
    cvt_f32_bf16<<<512,  256, 0, stream>>>(Wk, wkb, DM * DM);
    cvt_f32_bf16<<<512,  256, 0, stream>>>(Wv, wvb, DM * DM);
    cvt_f32_bf16<<<512,  256, 0, stream>>>(Wo, wob, DM * DM);

    proj_qkv<<<dim3(DM / 128, MROWS / 128, 3), 256, 0, stream>>>(
        xb, wqb, wkb, wvb, qh, kh, vt);

    attn<<<dim3(SEQ / 64, BSZ * NH), 128, 0, stream>>>(qh, kh, vt, ctx);

    out_proj<<<dim3(DM / 128, MROWS / 128), 256, 0, stream>>>(ctx, wob, out);
}
